// SparseRandomAttention_1864015807018
// MI455X (gfx1250) — compile-verified, matrix operand order checked by reading, not run on hardware
//
#include <hip/hip_runtime.h>
#include <hip/hip_bf16.h>

// ---------------------------------------------------------------------------
// SparseRandomAttention on gfx1250 (MI455X), bf16 WMMA pipeline + TDM staging.
//   B=512, S=256, D=1024, H=16, dh=64
// Pipeline:
//   1) cast x, Wq, Wk, Wv, Wo to bf16 (workspace)
//   2) proj_gemm: Q,K -> (B,H,S,64) bf16 ; V -> transposed (B,1024,256) bf16
//      A-tiles double-buffered into LDS via tensor_load_to_lds (TDM).
//   3) attn_kernel: per (b, 16-row tile): scores via WMMA, masked softmax in
//      registers, head-mean via LDS ds_add_f32, write avg_attention (f32 out),
//      then ctx = avg @ V via WMMA -> (B*S,1024) bf16
//   4) out_gemm: out = ctx @ Wo^T + bo -> f32 output (TDM-staged A)
// ---------------------------------------------------------------------------

typedef __attribute__((ext_vector_type(16))) __bf16 v16bf;
typedef __attribute__((ext_vector_type(8)))  __bf16 v8bf;
typedef __attribute__((ext_vector_type(4)))  __bf16 v4bf;
typedef __attribute__((ext_vector_type(8)))  float  v8f;
typedef __attribute__((ext_vector_type(4)))  float  v4f;

#if defined(__gfx1250__) && __has_builtin(__builtin_amdgcn_tensor_load_to_lds)
#define USE_TDM 1
#else
#define USE_TDM 0
#endif

#if USE_TDM
typedef __attribute__((ext_vector_type(4))) unsigned int v4u;
typedef __attribute__((ext_vector_type(8))) int         v8i;
typedef __attribute__((ext_vector_type(4))) int         v4i;

// TDM load of a 16(rows) x 128(cols) bf16 tile from a row-major matrix with a
// 1024-element row stride into LDS. D# groups per CDNA5 ISA:
//  g0: count=1 | lds_addr | global_addr[56:0] | type=2
//  g1: data_size=2B, tensor_dim0=1024, tensor_dim1=131072,
//      tile_dim0=128, tile_dim1=16, tensor_dim0_stride=1024
__device__ __forceinline__ void tdm_load_16x128(unsigned lds_addr,
                                                const __bf16* gsrc) {
  const unsigned long long ga = (unsigned long long)(size_t)gsrc;
  v4u g0;
  g0[0] = 1u;                                               // count=1, valid
  g0[1] = lds_addr;                                         // LDS byte addr
  g0[2] = (unsigned)ga;                                     // gaddr[31:0]
  g0[3] = ((unsigned)(ga >> 32) & 0x1FFFFFFu) | 0x80000000u; // gaddr[56:32]|type=2
  v8i g1;
  g1[0] = 0x00010000;                  // data_size code 1 (2 bytes)
  g1[1] = (int)(1024u << 16);          // tensor_dim0[15:0]=1024
  g1[2] = 0;                           // tdim0[31:16]=0 | tdim1[15:0]=0
  g1[3] = (int)(2u | (128u << 16));    // tdim1[31:16]=2 (131072) | tile_dim0=128
  g1[4] = 16;                          // tile_dim1=16, tile_dim2=0
  g1[5] = 1024;                        // tensor_dim0_stride[31:0]=1024
  g1[6] = 0;
  g1[7] = 0;
  const v4i z4 = {};
  const v8i z8 = {};
  // clang-23 / therock signature: (v4u, v8i, v4i, v4i, v8i, i32 cpol)
  __builtin_amdgcn_tensor_load_to_lds(g0, g1, z4, z4, z8, 0);
}
#endif

// Per-lane tile fragment loader matching the CDNA5 16-bit WMMA A/B layout:
// lane holds 8 contiguous bf16 at +0 and 8 contiguous bf16 at +16 elements.
__device__ __forceinline__ v16bf load_tile16(const __bf16* p) {
  v8bf lo = *(const v8bf*)(p);
  v8bf hi = *(const v8bf*)(p + 16);
  v16bf r;
#pragma unroll
  for (int i = 0; i < 8; ++i) { r[i] = lo[i]; r[i + 8] = hi[i]; }
  return r;
}

__device__ __forceinline__ v8f wmma_bf16(v16bf a, v16bf b, v8f c) {
  return __builtin_amdgcn_wmma_f32_16x16x32_bf16(false, a, false, b, (short)0,
                                                 c, false, false);
}

// ---------------------------------------------------------------------------
__global__ void cast_bf16_kernel(const float* __restrict__ src,
                                 __bf16* __restrict__ dst, long n) {
  long i = (long)blockIdx.x * blockDim.x + threadIdx.x;
  const long stride = (long)gridDim.x * blockDim.x;
  for (; i < n; i += stride) dst[i] = (__bf16)src[i];
}

// ---------------------------------------------------------------------------
// out tiles of X(131072 x 1024) @ W^T, W row-major (N=1024,K=1024).
// mode 0: scatter into (B,H,S,64) bf16 layout (Q/K).
// mode 1: scatter into transposed (B,1024,256) bf16 layout (V).
__global__ void __launch_bounds__(128)
proj_gemm(const __bf16* __restrict__ X, const __bf16* __restrict__ W,
          const float* __restrict__ bias, __bf16* __restrict__ out, int mode) {
  __shared__ __attribute__((aligned(16))) __bf16 sA[2][16 * 128];
  const int tid  = threadIdx.x;
  const int wave = tid >> 5, lane = tid & 31;
  const int rowbase = blockIdx.x * 16;
  const int colbase = blockIdx.y * 256 + wave * 64;
  const int lrow = lane & 15;
  const int koff = (lane >> 4) * 8;
  const __bf16* arow = X + (size_t)rowbase * 1024;

#if USE_TDM
  if (wave == 0) tdm_load_16x128((unsigned)(size_t)(void*)&sA[0][0], arow);
#endif

  v8f acc[4] = {};
#pragma unroll 1
  for (int kt0 = 0; kt0 < 8; ++kt0) {
    const int cur = kt0 & 1;
#if USE_TDM
    if (wave == 0) {
      if (kt0 < 7) {
        // prefetch next chunk into the other buffer, wait only for current
        tdm_load_16x128((unsigned)(size_t)(void*)&sA[cur ^ 1][0],
                        arow + (kt0 + 1) * 128);
        __builtin_amdgcn_s_wait_tensorcnt(1);
      } else {
        __builtin_amdgcn_s_wait_tensorcnt(0);
      }
    }
#else
    for (int i = tid; i < 256; i += 128) {
      const int r = i >> 4, c = (i & 15) * 8;
      *(v8bf*)&sA[cur][r * 128 + c] =
          *(const v8bf*)&arow[(size_t)r * 1024 + kt0 * 128 + c];
    }
#endif
    __syncthreads();  // publish sA[cur]
#pragma unroll
    for (int kk = 0; kk < 4; ++kk) {
      const int k = kt0 * 128 + kk * 32;
      const v16bf a = load_tile16(&sA[cur][lrow * 128 + kk * 32 + koff]);
      v16bf bm[4];
#pragma unroll
      for (int nt = 0; nt < 4; ++nt)
        bm[nt] = load_tile16(W + (size_t)(colbase + nt * 16 + lrow) * 1024 +
                             k + koff);
#pragma unroll
      for (int nt = 0; nt < 4; ++nt) acc[nt] = wmma_bf16(a, bm[nt], acc[nt]);
    }
    __syncthreads();  // all reads of sA[cur] done before it is re-filled
  }

#pragma unroll
  for (int nt = 0; nt < 4; ++nt) {
    const int n  = colbase + nt * 16 + lrow;
    const float bn = bias[n];
#pragma unroll
    for (int v = 0; v < 8; ++v) {
      const int m = ((lane >> 4) << 3) + v;   // D layout: vgpr v -> row m
      const int r = rowbase + m;
      const float val = acc[nt][v] + bn;
      if (mode == 0) {
        // (B,H,S,64): b=r>>8, s=r&255, h=n>>6, e=n&63
        out[((((size_t)(r >> 8)) * 16 + (n >> 6)) * 256 + (r & 255)) * 64 +
            (n & 63)] = (__bf16)val;
      } else {
        // (B,1024,256): transposed V
        out[(((size_t)(r >> 8)) * 1024 + n) * 256 + (r & 255)] = (__bf16)val;
      }
    }
  }
}

// ---------------------------------------------------------------------------
// One block per (batch b, 16-row tile). 4 waves; each wave owns 4 heads for
// the scores/softmax phase, then a 256-column strip for the ctx GEMM.
__global__ void __launch_bounds__(128)
attn_kernel(const __bf16* __restrict__ Qb,     // (B,16,256,64)
            const __bf16* __restrict__ Kb,     // (B,16,256,64)
            const __bf16* __restrict__ vT,     // (B,1024,256)
            const unsigned char* __restrict__ mask, // (256,256) bool
            float* __restrict__ avg_out,       // (B,256,256)
            __bf16* __restrict__ ctx) {        // (B*256,1024)
  __shared__ unsigned int smask[16][8];
  __shared__ float savg[16][256];
  __shared__ __attribute__((aligned(16))) __bf16 sattn[16 * 256];

  const int tid  = threadIdx.x;
  const int wave = tid >> 5, lane = tid & 31;
  const int b       = blockIdx.x >> 4;
  const int rowbase = (blockIdx.x & 15) * 16;
  const int lrow = lane & 15;
  const int koff = (lane >> 4) * 8;

  for (int i = tid; i < 16 * 256; i += 128) ((float*)savg)[i] = 0.0f;
  {
    // build 256-bit row masks: dword loads + byte->bit multiply compress
    const int r = tid >> 3, w = tid & 7;
    const unsigned* mrow =
        (const unsigned*)(mask + (size_t)(rowbase + r) * 256) + w * 8;
    unsigned bits = 0;
#pragma unroll
    for (int j = 0; j < 8; ++j) {
      const unsigned nib =
          (((mrow[j] & 0x01010101u) * 0x01020408u) >> 24) & 0xFu;
      bits |= nib << (j * 4);
    }
    smask[r][w] = bits;
  }
  __syncthreads();

  // ---- scores + masked softmax + head-mean accumulation -------------------
  for (int h = wave; h < 16; h += 4) {
    const __bf16* qp =
        Qb + (((size_t)b * 16 + h) * 256 + rowbase + lrow) * 64 + koff;
    const v16bf a0 = load_tile16(qp);
    const v16bf a1 = load_tile16(qp + 32);
    const __bf16* kbase =
        Kb + (((size_t)b * 16 + h) * 256 + lrow) * 64 + koff;

    v8f sc[16];
#pragma unroll
    for (int t2 = 0; t2 < 8; ++t2) {
      // two 16-col tiles per trip: 4 b128-pairs in flight before the WMMAs
      const __bf16* kp0 = kbase + (size_t)(t2 * 32) * 64;
      const __bf16* kp1 = kp0 + 16 * 64;
      const v16bf b00 = load_tile16(kp0);
      const v16bf b01 = load_tile16(kp0 + 32);
      const v16bf b10 = load_tile16(kp1);
      const v16bf b11 = load_tile16(kp1 + 32);
      v8f c0 = {}, c1 = {};
      c0 = wmma_bf16(a0, b00, c0);
      c1 = wmma_bf16(a0, b10, c1);
      c0 = wmma_bf16(a1, b01, c0);
      c1 = wmma_bf16(a1, b11, c1);
      sc[2 * t2]     = c0;
      sc[2 * t2 + 1] = c1;
    }

    // scale by 1/sqrt(64) and apply the sparse mask
#pragma unroll
    for (int t = 0; t < 16; ++t) {
      const unsigned bitpos = ((t & 1) << 4) | (unsigned)lrow;
      const int w = t >> 1;
#pragma unroll
      for (int v = 0; v < 8; ++v) {
        const int m = ((lane >> 4) << 3) + v;
        const bool on = (smask[m][w] >> bitpos) & 1u;
        sc[t][v] = on ? sc[t][v] * 0.125f : -1.0e30f;
      }
    }

    // row-wise max over 16 tiles (in lane) then across the 16-lane half
    float mx[8], sm[8];
#pragma unroll
    for (int v = 0; v < 8; ++v) mx[v] = -1.0e30f;
#pragma unroll
    for (int t = 0; t < 16; ++t)
#pragma unroll
      for (int v = 0; v < 8; ++v) mx[v] = fmaxf(mx[v], sc[t][v]);
#pragma unroll
    for (int off = 1; off < 16; off <<= 1)
#pragma unroll
      for (int v = 0; v < 8; ++v)
        mx[v] = fmaxf(mx[v], __shfl_xor(mx[v], off, 16));

#pragma unroll
    for (int v = 0; v < 8; ++v) sm[v] = 0.0f;
#pragma unroll
    for (int t = 0; t < 16; ++t)
#pragma unroll
      for (int v = 0; v < 8; ++v) {
        const float e = __expf(sc[t][v] - mx[v]);
        sc[t][v] = e;
        sm[v] += e;
      }
#pragma unroll
    for (int off = 1; off < 16; off <<= 1)
#pragma unroll
      for (int v = 0; v < 8; ++v) sm[v] += __shfl_xor(sm[v], off, 16);
#pragma unroll
    for (int v = 0; v < 8; ++v) sm[v] = 1.0f / sm[v];

    // accumulate softmax probabilities into shared head-mean buffer
#pragma unroll
    for (int t = 0; t < 16; ++t)
#pragma unroll
      for (int v = 0; v < 8; ++v) {
        const int m = ((lane >> 4) << 3) + v;
        atomicAdd(&savg[m][t * 16 + lrow], sc[t][v] * sm[v]);
      }
  }
  __syncthreads();

  // ---- emit avg_attention (f32 output #2) + stage bf16 copy for ctx GEMM --
  for (int i = tid; i < 1024; i += 128) {
    const int r = i >> 6, c = (i & 63) * 4;
    v4f val = *(const v4f*)&savg[r][c];
    val *= 0.0625f;  // mean over 16 heads
    *(v4f*)&avg_out[((size_t)b * 256 + rowbase + r) * 256 + c] = val;
    v4bf pk;
#pragma unroll
    for (int j = 0; j < 4; ++j) pk[j] = (__bf16)val[j];
    *(v4bf*)&sattn[r * 256 + c] = pk;
  }
  __syncthreads();

  // ---- ctx tile (16 x 1024) = avg (16 x 256) @ V (256 x 1024) -------------
  v8f cc[16] = {};
#pragma unroll 1
  for (int kt = 0; kt < 8; ++kt) {
    const int k = kt * 32;
    const v16bf a = load_tile16(&sattn[lrow * 256 + k + koff]);
#pragma unroll
    for (int ng = 0; ng < 4; ++ng) {
      v16bf bm[4];
#pragma unroll
      for (int j = 0; j < 4; ++j) {
        const int n = wave * 256 + (ng * 4 + j) * 16 + lrow;
        bm[j] = load_tile16(vT + ((size_t)b * 1024 + n) * 256 + k + koff);
      }
#pragma unroll
      for (int j = 0; j < 4; ++j)
        cc[ng * 4 + j] = wmma_bf16(a, bm[j], cc[ng * 4 + j]);
    }
  }
#pragma unroll
  for (int nt = 0; nt < 16; ++nt) {
    const int n = wave * 256 + nt * 16 + lrow;
#pragma unroll
    for (int v = 0; v < 8; ++v) {
      const int m = ((lane >> 4) << 3) + v;
      ctx[((size_t)b * 256 + rowbase + m) * 1024 + n] = (__bf16)cc[nt][v];
    }
  }
}

// ---------------------------------------------------------------------------
// out(131072 x 1024) f32 = ctx @ Wo^T + bo
__global__ void __launch_bounds__(128)
out_gemm(const __bf16* __restrict__ ctx, const __bf16* __restrict__ Wo,
         const float* __restrict__ bo, float* __restrict__ out) {
  __shared__ __attribute__((aligned(16))) __bf16 sA[2][16 * 128];
  const int tid  = threadIdx.x;
  const int wave = tid >> 5, lane = tid & 31;
  const int rowbase = blockIdx.x * 16;
  const int colbase = blockIdx.y * 256 + wave * 64;
  const int lrow = lane & 15;
  const int koff = (lane >> 4) * 8;
  const __bf16* arow = ctx + (size_t)rowbase * 1024;

#if USE_TDM
  if (wave == 0) tdm_load_16x128((unsigned)(size_t)(void*)&sA[0][0], arow);
#endif

  v8f acc[4] = {};
#pragma unroll 1
  for (int kt0 = 0; kt0 < 8; ++kt0) {
    const int cur = kt0 & 1;
#if USE_TDM
    if (wave == 0) {
      if (kt0 < 7) {
        tdm_load_16x128((unsigned)(size_t)(void*)&sA[cur ^ 1][0],
                        arow + (kt0 + 1) * 128);
        __builtin_amdgcn_s_wait_tensorcnt(1);
      } else {
        __builtin_amdgcn_s_wait_tensorcnt(0);
      }
    }
#else
    for (int i = tid; i < 256; i += 128) {
      const int r = i >> 4, c = (i & 15) * 8;
      *(v8bf*)&sA[cur][r * 128 + c] =
          *(const v8bf*)&arow[(size_t)r * 1024 + kt0 * 128 + c];
    }
#endif
    __syncthreads();
#pragma unroll
    for (int kk = 0; kk < 4; ++kk) {
      const int k = kt0 * 128 + kk * 32;
      const v16bf a = load_tile16(&sA[cur][lrow * 128 + kk * 32 + koff]);
      v16bf bm[4];
#pragma unroll
      for (int nt = 0; nt < 4; ++nt)
        bm[nt] = load_tile16(Wo + (size_t)(colbase + nt * 16 + lrow) * 1024 +
                             k + koff);
#pragma unroll
      for (int nt = 0; nt < 4; ++nt) acc[nt] = wmma_bf16(a, bm[nt], acc[nt]);
    }
    __syncthreads();
  }
#pragma unroll
  for (int nt = 0; nt < 4; ++nt) {
    const int n = colbase + nt * 16 + lrow;
    const float bn = bo[n];
#pragma unroll
    for (int v = 0; v < 8; ++v) {
      const int m = ((lane >> 4) << 3) + v;
      out[(size_t)(rowbase + m) * 1024 + n] = acc[nt][v] + bn;
    }
  }
}

// ---------------------------------------------------------------------------
extern "C" void kernel_launch(void* const* d_in, const int* in_sizes, int n_in,
                              void* d_out, int out_size, void* d_ws,
                              size_t ws_size, hipStream_t stream) {
  const float* x  = (const float*)d_in[0];
  const float* Wq = (const float*)d_in[1];
  const float* bq = (const float*)d_in[2];
  const float* Wk = (const float*)d_in[3];
  const float* bk = (const float*)d_in[4];
  const float* Wv = (const float*)d_in[5];
  const float* bv = (const float*)d_in[6];
  const float* Wo = (const float*)d_in[7];
  const float* bo = (const float*)d_in[8];
  const unsigned char* mask = (const unsigned char*)d_in[9];

  const long long BIG = 268435456LL;  // 512*256*1024 * 2 bytes
  char* ws = (char*)d_ws;
  __bf16* xb   = (__bf16*)(ws + 0 * BIG);  // x bf16
  __bf16* qb   = (__bf16*)(ws + 1 * BIG);  // Q (B,H,S,64)
  __bf16* kbuf = (__bf16*)(ws + 2 * BIG);  // K (B,H,S,64)
  __bf16* vt   = (__bf16*)(ws + 3 * BIG);  // V^T (B,1024,256)
  __bf16* ctx  = (__bf16*)(ws + 4 * BIG);  // avg@V (B*S,1024)
  __bf16* wqb  = (__bf16*)(ws + 5 * BIG);
  __bf16* wkb  = (__bf16*)(ws + 5 * BIG + 2097152LL);
  __bf16* wvb  = (__bf16*)(ws + 5 * BIG + 2 * 2097152LL);
  __bf16* wob  = (__bf16*)(ws + 5 * BIG + 3 * 2097152LL);

  float* out_main = (float*)d_out;                    // (B,S,1024)
  float* avg_out  = (float*)d_out + 134217728LL;      // (B,256,256)

  cast_bf16_kernel<<<4096, 256, 0, stream>>>(x, xb, 134217728L);
  cast_bf16_kernel<<<256, 256, 0, stream>>>(Wq, wqb, 1048576L);
  cast_bf16_kernel<<<256, 256, 0, stream>>>(Wk, wkb, 1048576L);
  cast_bf16_kernel<<<256, 256, 0, stream>>>(Wv, wvb, 1048576L);
  cast_bf16_kernel<<<256, 256, 0, stream>>>(Wo, wob, 1048576L);

  dim3 gproj(8192, 4);
  proj_gemm<<<gproj, 128, 0, stream>>>(xb, wqb, bq, qb, 0);
  proj_gemm<<<gproj, 128, 0, stream>>>(xb, wkb, bk, kbuf, 0);
  proj_gemm<<<gproj, 128, 0, stream>>>(xb, wvb, bv, vt, 1);

  attn_kernel<<<8192, 128, 0, stream>>>(qb, kbuf, vt, mask, avg_out, ctx);

  out_gemm<<<dim3(8192, 4), 128, 0, stream>>>(ctx, wob, bo, out_main);
}
